// PlayLSTM_7602092114093
// MI455X (gfx1250) — matrix operand order repaired and checked by reading, not compiled, not run on hardware
//
#include <hip/hip_runtime.h>
#include <hip/hip_bf16.h>

typedef __bf16 bf16_t;
typedef __attribute__((ext_vector_type(16))) __bf16 v16bf;
typedef __attribute__((ext_vector_type(8)))  __bf16 v8bf;
typedef __attribute__((ext_vector_type(8)))  float  v8f;

#define B_  128
#define T_  1024
#define D_  256
#define H_  256
#define G4_ 1024   // 4*H
#define K_  512    // D + H (fused [x_t ; h] K dimension)
#define NBLK 32    // 32 blocks x 4 waves = 128 tiles of 16x16 over [B,H]

// ---------------------------------------------------------------- utilities

__device__ __forceinline__ float fast_sigmoid(float x) {
    return 1.0f / (1.0f + __expf(-x));
}

// Hierarchical grid barrier: waves aggregate via workgroup barrier, then one
// arrival per block on a monotonic agent-scope counter (32 arrivals/step).
__device__ __forceinline__ void grid_barrier(unsigned* cnt, unsigned barNum) {
    __threadfence();        // flush this wave's h writes to device scope
    __syncthreads();        // all 4 waves' fences done
    if (threadIdx.x == 0) {
        __hip_atomic_fetch_add(cnt, 1u, __ATOMIC_RELEASE, __HIP_MEMORY_SCOPE_AGENT);
    }
    const unsigned target = (unsigned)NBLK * barNum;
    while (__hip_atomic_load(cnt, __ATOMIC_ACQUIRE, __HIP_MEMORY_SCOPE_AGENT) < target) {
        __builtin_amdgcn_s_sleep(2);
    }
    __threadfence();        // acquire: drop stale near-cache h lines
}

// ---------------------------------------------------------------- prep kernels

__global__ void lstm_init_counter(unsigned* cnt) {
    if (blockIdx.x == 0 && threadIdx.x == 0) *cnt = 0u;
}

// Wcat[n][k]: n in [0,4H), k in [0,512): k<256 -> Wi[k][n], else Wh[k-256][n].
// Row-major over n so each WMMA B-lane reads two 8-wide K chunks (K 8h..8h+7, 16+8h..16+8h+7).
__global__ void lstm_prep_w(const float* __restrict__ Wi,
                            const float* __restrict__ Wh,
                            bf16_t* __restrict__ Wcat) {
    int i = blockIdx.x * blockDim.x + threadIdx.x;
    if (i >= G4_ * K_) return;
    int n = i / K_;
    int k = i - n * K_;
    float v = (k < D_) ? Wi[(size_t)k * G4_ + n] : Wh[(size_t)(k - D_) * G4_ + n];
    Wcat[i] = (bf16_t)v;
}

// fp32 x -> bf16 x, same [B,T,D] layout; 8 elements per thread, b128 stores.
__global__ void lstm_prep_x(const float* __restrict__ x, bf16_t* __restrict__ xb) {
    size_t i = ((size_t)blockIdx.x * blockDim.x + threadIdx.x) * 8;
    if (i >= (size_t)B_ * T_ * D_) return;
    float4 v0 = *(const float4*)(x + i);
    float4 v1 = *(const float4*)(x + i + 4);
    v8bf o;
    o[0] = (__bf16)v0.x; o[1] = (__bf16)v0.y; o[2] = (__bf16)v0.z; o[3] = (__bf16)v0.w;
    o[4] = (__bf16)v1.x; o[5] = (__bf16)v1.y; o[6] = (__bf16)v1.z; o[7] = (__bf16)v1.w;
    *(v8bf*)(xb + i) = o;
}

// ---------------------------------------------------------------- persistent LSTM

// Load one 16x32 bf16 A-fragment (ISA layout: lanes 0-15 hold K 0-7/16-23,
// lanes 16-31 hold K 8-15/24-31); koff8 = (lane>=16)*8.
__device__ __forceinline__ v16bf ld_a32(const bf16_t* aBase, int koff8) {
    v8bf lo = *(const v8bf*)(aBase + koff8);
    v8bf hi = *(const v8bf*)(aBase + 16 + koff8);
    return __builtin_shufflevector(lo, hi,
        0, 1, 2, 3, 4, 5, 6, 7, 8, 9, 10, 11, 12, 13, 14, 15);
}

__global__ __launch_bounds__(128, 1) void lstm_persist(
    const bf16_t* __restrict__ xb,    // [B,T,D] bf16
    const bf16_t* __restrict__ Wcat,  // [4H][K_] bf16
    const float*  __restrict__ bh,    // [4H]
    bf16_t*       __restrict__ hbuf,  // 2 * B*H bf16 (double-buffered h)
    float*        __restrict__ out,   // outputs [B,T,H] ++ c_fin [B,H] ++ h_fin [B,H]
    unsigned*     __restrict__ barCnt)
{
    const int lane = threadIdx.x & 31;
    const int wave = threadIdx.x >> 5;     // 0..3
    const int bid  = blockIdx.x;           // 0..31
    const int mt   = bid & 7;              // batch tile (shared by all 4 waves)
    const int nt   = (bid >> 3) * 4 + wave; // hidden tile 0..15
    const int m0   = mt * 16;
    const int n0   = nt * 16;
    const int ncol = lane & 15;
    const int hi   = lane >> 4;
    const int row  = m0 + ncol;            // A-matrix row for this lane
    const int koff8  = hi * 8;             // A chunk select
    const int koff16 = hi * 16;            // unused (B fragments load with koff8)

    // ---- register-resident weights: 16 K-chunks x 4 gates (512 VGPRs) ----
    v16bf bfrag[16][4];
#pragma unroll
    for (int kk = 0; kk < 16; ++kk)
#pragma unroll
        for (int g = 0; g < 4; ++g)
            bfrag[kk][g] = ld_a32(
                Wcat + (size_t)(g * H_ + n0 + ncol) * K_ + kk * 32, koff8);

    // Per-lane gate biases
    float bias[4];
#pragma unroll
    for (int g = 0; g < 4; ++g) bias[g] = bh[g * H_ + n0 + ncol];

    // Cell state lives in registers for all 1024 steps. C/D layout:
    // element r of the f32 accumulator is (M = r + 8*hi, N = ncol).
    float creg[8];
    float hlast[8];
#pragma unroll
    for (int r = 0; r < 8; ++r) { creg[r] = 0.0f; hlast[r] = 0.0f; }

    // Zero h0 (parity 0): each wave zeroes exactly its own tile.
#pragma unroll
    for (int r = 0; r < 8; ++r) {
        int m = m0 + r + hi * 8;
        hbuf[(size_t)m * H_ + n0 + ncol] = (bf16_t)0.0f;
    }

    // Prefetch x A-fragments for t = 0 (independent of h).
    const bf16_t* aX0 = xb + (size_t)row * T_ * D_;
    v16bf xfrag[8];
#pragma unroll
    for (int kk = 0; kk < 8; ++kk) xfrag[kk] = ld_a32(aX0 + kk * 32, koff8);

    unsigned barNum = 1;
    grid_barrier(barCnt, barNum++);

    for (int t = 0; t < T_; ++t) {
        const bf16_t* hin  = hbuf + (size_t)(t & 1) * (B_ * H_);
        bf16_t*       hout = hbuf + (size_t)((t + 1) & 1) * (B_ * H_);

        v8f acc[4];
#pragma unroll
        for (int g = 0; g < 4; ++g)
#pragma unroll
            for (int r = 0; r < 8; ++r) acc[g][r] = bias[g];

        // x-side: prefetched fragments (K 0..255)
#pragma unroll
        for (int kk = 0; kk < 8; ++kk)
#pragma unroll
            for (int g = 0; g < 4; ++g)
                acc[g] = __builtin_amdgcn_wmma_f32_16x16x32_bf16(
                    false, xfrag[kk], false, bfrag[kk][g], (short)0, acc[g],
                    false, false);

        // h-side: fresh loads (K 256..511)
        const bf16_t* aH = hin + (size_t)row * H_;
#pragma unroll
        for (int kk = 0; kk < 8; ++kk) {
            v16bf a = ld_a32(aH + kk * 32, koff8);
#pragma unroll
            for (int g = 0; g < 4; ++g)
                acc[g] = __builtin_amdgcn_wmma_f32_16x16x32_bf16(
                    false, a, false, bfrag[8 + kk][g], (short)0, acc[g],
                    false, false);
        }

        // Element-wise cell update; fp32 throughout.
#pragma unroll
        for (int r = 0; r < 8; ++r) {
            const int m = m0 + r + hi * 8;
            const int n = n0 + ncol;
            float iv = fast_sigmoid(acc[0][r]);
            float fv = fast_sigmoid(acc[1][r]);
            float gv = tanhf(acc[2][r]);
            float ov = fast_sigmoid(acc[3][r]);
            float cn = fv * creg[r] + iv * gv;
            creg[r] = cn;
            float hn = ov * tanhf(cn);
            hlast[r] = hn;
            out[((size_t)m * T_ + t) * H_ + n] = hn;
            hout[(size_t)m * H_ + n] = (bf16_t)hn;
        }

        // Prefetch next step's x fragments so they land during the barrier spin.
        if (t + 1 < T_) {
            const bf16_t* aX = aX0 + (size_t)(t + 1) * D_;
#pragma unroll
            for (int kk = 0; kk < 8; ++kk)
                xfrag[kk] = ld_a32(aX + kk * 32, koff8);
        }

        if (t != T_ - 1) grid_barrier(barCnt, barNum++);
    }

    // Final carry: c_fin then h_fin appended after outputs.
    float* cfin = out + (size_t)B_ * T_ * H_;
    float* hfin = cfin + (size_t)B_ * H_;
#pragma unroll
    for (int r = 0; r < 8; ++r) {
        const int m = m0 + r + hi * 8;
        const int n = n0 + ncol;
        cfin[(size_t)m * H_ + n] = creg[r];
        hfin[(size_t)m * H_ + n] = hlast[r];
    }
}

// ---------------------------------------------------------------- launch

extern "C" void kernel_launch(void* const* d_in, const int* in_sizes, int n_in,
                              void* d_out, int out_size, void* d_ws, size_t ws_size,
                              hipStream_t stream) {
    const float* x  = (const float*)d_in[0];   // [B,T,D]
    const float* Wi = (const float*)d_in[1];   // [D,4H]
    const float* Wh = (const float*)d_in[2];   // [H,4H]
    const float* bh = (const float*)d_in[3];   // [4H]
    float* out = (float*)d_out;

    char* ws = (char*)d_ws;
    size_t off = 0;
    bf16_t* xb = (bf16_t*)(ws + off);   off += (size_t)B_ * T_ * D_ * sizeof(bf16_t);
    bf16_t* Wcat = (bf16_t*)(ws + off); off += (size_t)G4_ * K_ * sizeof(bf16_t);
    bf16_t* hbuf = (bf16_t*)(ws + off); off += (size_t)2 * B_ * H_ * sizeof(bf16_t);
    off = (off + 255) & ~(size_t)255;
    unsigned* barCnt = (unsigned*)(ws + off);

    lstm_init_counter<<<1, 32, 0, stream>>>(barCnt);

    lstm_prep_w<<<(G4_ * K_ + 255) / 256, 256, 0, stream>>>(Wi, Wh, Wcat);

    size_t nx8 = (size_t)B_ * T_ * D_ / 8;
    lstm_prep_x<<<(unsigned)((nx8 + 255) / 256), 256, 0, stream>>>(x, xb);

    lstm_persist<<<NBLK, 128, 0, stream>>>(xb, Wcat, bh, hbuf, out, barCnt);
}
